// Model_29789893165151
// MI455X (gfx1250) — matrix operand order repaired and checked by reading, compiled, not run on hardware
//
#include <hip/hip_runtime.h>
#include <math.h>

typedef __attribute__((ext_vector_type(16))) __bf16 v16bf;
typedef __attribute__((ext_vector_type(8)))  __bf16 v8bf;
typedef __attribute__((ext_vector_type(8)))  float  v8f;

#define LOOPN 2048
#define BATCHN 2048
#define EPSV 1e-5f

__device__ __forceinline__ float swishf(float x) {
    return x / (1.0f + __expf(-x));
}

// 3x3 conv, zero padding, on a 4x4 tile held in registers.
__device__ __forceinline__ void conv4x4(const float* xr, const float* w, float* o) {
#pragma unroll
    for (int i = 0; i < 4; ++i) {
#pragma unroll
        for (int j = 0; j < 4; ++j) {
            float s = 0.0f;
#pragma unroll
            for (int di = -1; di <= 1; ++di) {
#pragma unroll
                for (int dj = -1; dj <= 1; ++dj) {
                    int ii = i + di, jj = j + dj;
                    if (ii >= 0 && ii < 4 && jj >= 0 && jj < 4)
                        s += w[(di + 1) * 3 + (dj + 1)] * xr[ii * 4 + jj];
                }
            }
            o[i * 4 + j] = s;
        }
    }
}

// One block per channel: conv+swish over all 2048 batches, reduce to mean/rstd.
__global__ __launch_bounds__(256) void conv_stats(
    const float* __restrict__ xa, const float* __restrict__ conv_w,
    const float* __restrict__ conv_b, float* __restrict__ mean,
    float* __restrict__ rstd) {
    int c = blockIdx.x;
    __shared__ float sw[9];
    if (threadIdx.x < 9) sw[threadIdx.x] = conv_w[c * 9 + threadIdx.x];
    __syncthreads();
    float w[9];
#pragma unroll
    for (int i = 0; i < 9; ++i) w[i] = sw[i];
    float cb = conv_b[c];

    float s = 0.0f, sq = 0.0f;
    for (int n = threadIdx.x; n < BATCHN; n += 256) {
        float xr[16];
        const float* xp = xa + n * 16;
#pragma unroll
        for (int i = 0; i < 16; ++i) xr[i] = xp[i];
        float o[16];
        conv4x4(xr, w, o);
#pragma unroll
        for (int p = 0; p < 16; ++p) {
            float v = swishf(o[p] + cb);
            s += v; sq += v * v;
        }
    }
    __shared__ float rs[256], rq[256];
    rs[threadIdx.x] = s; rq[threadIdx.x] = sq;
    __syncthreads();
    for (int st = 128; st > 0; st >>= 1) {
        if (threadIdx.x < st) {
            rs[threadIdx.x] += rs[threadIdx.x + st];
            rq[threadIdx.x] += rq[threadIdx.x + st];
        }
        __syncthreads();
    }
    if (threadIdx.x == 0) {
        const float inv = 1.0f / (float)(BATCHN * 16);
        float m = rs[0] * inv;
        float v = rq[0] * inv - m * m;
        mean[c] = m;
        rstd[c] = rsqrtf(v + EPSV);
    }
}

// AdaptiveAvgPool2d((2,2)) on 4x4 per batch row -> bp[c][4].
__global__ void pool_b(const float* __restrict__ xa, float* __restrict__ bp) {
    int c = blockIdx.x * blockDim.x + threadIdx.x;
    if (c >= LOOPN) return;
    const float* x = xa + c * 16;
    bp[c * 4 + 0] = 0.25f * (x[0] + x[1] + x[4] + x[5]);
    bp[c * 4 + 1] = 0.25f * (x[2] + x[3] + x[6] + x[7]);
    bp[c * 4 + 2] = 0.25f * (x[8] + x[9] + x[12] + x[13]);
    bp[c * 4 + 3] = 0.25f * (x[10] + x[11] + x[14] + x[15]);
}

// Recompute conv+swish, apply BN affine and bp broadcast, emit bf16 z[n][c*16+p].
__global__ __launch_bounds__(256) void build_z(
    const float* __restrict__ xa, const float* __restrict__ conv_w,
    const float* __restrict__ conv_b, const float* __restrict__ bn_g,
    const float* __restrict__ bn_b, const float* __restrict__ mean,
    const float* __restrict__ rstd, const float* __restrict__ bp,
    __bf16* __restrict__ z) {
    int c = blockIdx.x;
    __shared__ float sw[9];
    if (threadIdx.x < 9) sw[threadIdx.x] = conv_w[c * 9 + threadIdx.x];
    __syncthreads();
    float w[9];
#pragma unroll
    for (int i = 0; i < 9; ++i) w[i] = sw[i];
    float cb = conv_b[c];
    float scale = rstd[c] * bn_g[c];
    float shift = bn_b[c] - mean[c] * scale;
    float b4[4];
#pragma unroll
    for (int j = 0; j < 4; ++j) b4[j] = bp[c * 4 + j];

    for (int n = threadIdx.x; n < BATCHN; n += 256) {
        float xr[16];
        const float* xp = xa + n * 16;
#pragma unroll
        for (int i = 0; i < 16; ++i) xr[i] = xp[i];
        float o[16];
        conv4x4(xr, w, o);
        v16bf out;
#pragma unroll
        for (int p = 0; p < 16; ++p) {
            float v = swishf(o[p] + cb);
            v = v * scale + shift;
            out[p] = (__bf16)(v * b4[p & 3]);
        }
        *(v16bf*)(z + (size_t)n * (LOOPN * 16) + c * 16) = out;
    }
}

// f32 -> bf16 with zero-padded K dimension: dst is rows x Kp, src rows x Ksrc.
__global__ void cvt_pad(const float* __restrict__ src, __bf16* __restrict__ dst,
                        long rows, long Ksrc, long Kp) {
    long i = (long)blockIdx.x * blockDim.x + threadIdx.x;
    long stride = (long)gridDim.x * blockDim.x;
    long n = rows * Kp;
    for (; i < n; i += stride) {
        long r = i / Kp, k = i - r * Kp;
        dst[i] = (k < Ksrc) ? (__bf16)src[r * Ksrc + k] : (__bf16)0.0f;
    }
}

// Row-wise LayerNorm(N), emit bf16 padded to Npad (zeros beyond N).
__global__ __launch_bounds__(256) void ln_rows(
    const float* __restrict__ h, const float* __restrict__ g,
    const float* __restrict__ b, __bf16* __restrict__ out, int N, int Npad,
    int ldh) {
    int row = blockIdx.x;
    const float* hp = h + (size_t)row * ldh;
    float s = 0.0f, sq = 0.0f;
    for (int i = threadIdx.x; i < N; i += 256) {
        float v = hp[i];
        s += v; sq += v * v;
    }
    __shared__ float rs[256], rq[256];
    rs[threadIdx.x] = s; rq[threadIdx.x] = sq;
    __syncthreads();
    for (int st = 128; st > 0; st >>= 1) {
        if (threadIdx.x < st) {
            rs[threadIdx.x] += rs[threadIdx.x + st];
            rq[threadIdx.x] += rq[threadIdx.x + st];
        }
        __syncthreads();
    }
    __shared__ float smu, srs;
    if (threadIdx.x == 0) {
        float mu = rs[0] / (float)N;
        float var = rq[0] / (float)N - mu * mu;
        smu = mu;
        srs = rsqrtf(var + EPSV);
    }
    __syncthreads();
    float mu = smu, rst = srs;
    for (int i = threadIdx.x; i < Npad; i += 256) {
        float v = (i < N) ? (hp[i] - mu) * rst * g[i] + b[i] : 0.0f;
        out[(size_t)row * Npad + i] = (__bf16)v;
    }
}

// Generic TN GEMM: C[M,Nstore] = act(A[M,K] * B[N,K]^T + bias), bf16 in,
// f32 WMMA accumulation. Caller guarantees:
//   K % 64 == 0 (zero-padded), M % 128 == 0, Nstore % 128 == 0, grid covers
//   exactly M/128 x Nstore/128 blocks.
// Block tile 128(M) x 128(N), BK=32, static double-buffered LDS; 8 waves in
// 4(M) x 2(N) grid; each wave computes a 32x64 patch = 8 WMMA per K-step.
// Out-of-range B rows are clamped (not zeroed): their products land only in
// pad columns (col >= N) which the epilogue overwrites with zeros.
#define BM 128
#define BN 128
#define BK 32
#define LDSROW 48  // row pitch in elems (96B): aligned b128 frags, bank spread

__global__ __launch_bounds__(256) void gemm_bf16(
    const __bf16* __restrict__ A, const __bf16* __restrict__ B,
    const float* __restrict__ bias, void* __restrict__ Cout, int M, int N,
    int Nstore, int K, int act, int out_bf16) {
    __shared__ __align__(16) __bf16 As[2][BM * LDSROW];   // 2 x 12 KB
    __shared__ __align__(16) __bf16 Bs[2][BN * LDSROW];   // 2 x 12 KB

    const int tid  = threadIdx.x;
    const int lane = tid & 31;
    const int wave = tid >> 5;   // 0..7
    const int wm   = wave >> 1;  // 4 waves along M (32 rows each)
    const int wn   = wave & 1;   // 2 waves along N (64 cols each)

    const int m0 = blockIdx.y * BM;
    const int n0 = blockIdx.x * BN;

    // A/B loaders: 4096 elems each, 16 per thread (two v8bf)
    const int lrow = tid >> 1;        // 0..127
    const int lcol = (tid & 1) * 16;  // 0,16

    // Clamp rows: A rows always valid (M % 128 == 0); B rows clamp to N-1.
    int ar = m0 + lrow; if (ar > M - 1) ar = M - 1;
    int br = n0 + lrow; if (br > N - 1) br = N - 1;
    const __bf16* aSrc = A + (size_t)ar * K + lcol;
    const __bf16* bSrc = B + (size_t)br * K + lcol;
    const int sOff = lrow * LDSROW + lcol;

    const int lh = lane >> 4;   // half-wave select
    const int lm = lane & 15;
    const int kha = lh * 8;     // A frag: K split 0..7 / 8..15 (+16)
    const int khb = lh * 8;     // B frag: K split 0..7 / 8..15 (+16)

    const int ra0 = (wm * 32 + lm) * LDSROW;
    const int ra1 = (wm * 32 + 16 + lm) * LDSROW;
    int rb[4];
#pragma unroll
    for (int u = 0; u < 4; ++u) rb[u] = (wn * 64 + u * 16 + lm) * LDSROW;

    v8f acc[2][4] = {{{}, {}, {}, {}}, {{}, {}, {}, {}}};
    v8bf aR0, aR1, bR0, bR1;

    auto loadRegs = [&](int kk) {
        aR0 = *(const v8bf*)(aSrc + kk);
        aR1 = *(const v8bf*)(aSrc + kk + 8);
        bR0 = *(const v8bf*)(bSrc + kk);
        bR1 = *(const v8bf*)(bSrc + kk + 8);
    };
    auto stage = [&](int buf) {
        *(v8bf*)&As[buf][sOff]     = aR0;
        *(v8bf*)&As[buf][sOff + 8] = aR1;
        *(v8bf*)&Bs[buf][sOff]     = bR0;
        *(v8bf*)&Bs[buf][sOff + 8] = bR1;
    };
    auto compute = [&](int buf) {
        const __bf16* Ab = As[buf];
        const __bf16* Bb = Bs[buf];
        v16bf a0, a1;
        ((v8bf*)&a0)[0] = *(const v8bf*)&Ab[ra0 + kha];
        ((v8bf*)&a0)[1] = *(const v8bf*)&Ab[ra0 + kha + 16];
        ((v8bf*)&a1)[0] = *(const v8bf*)&Ab[ra1 + kha];
        ((v8bf*)&a1)[1] = *(const v8bf*)&Ab[ra1 + kha + 16];
#pragma unroll
        for (int u = 0; u < 4; ++u) {
            v16bf bu;
            ((v8bf*)&bu)[0] = *(const v8bf*)&Bb[rb[u] + khb];
            ((v8bf*)&bu)[1] = *(const v8bf*)&Bb[rb[u] + khb + 16];
            acc[0][u] = __builtin_amdgcn_wmma_f32_16x16x32_bf16(
                false, a0, false, bu, (short)0, acc[0][u], false, false);
            acc[1][u] = __builtin_amdgcn_wmma_f32_16x16x32_bf16(
                false, a1, false, bu, (short)0, acc[1][u], false, false);
        }
    };

    // prologue: fill buffer 0
    loadRegs(0);
    stage(0);
    __syncthreads();

    for (int k0 = 0; k0 < K; k0 += 2 * BK) {
        // step A: prefetch k0+BK, compute on buf0, stage buf1
        loadRegs(k0 + BK);           // k0+BK <= K-BK always (K % 64 == 0)
        compute(0);
        stage(1);
        __syncthreads();
        // step B: prefetch k0+2BK (unless last), compute buf1, stage buf0
        const bool notLast = (k0 + 2 * BK) < K;
        if (notLast) loadRegs(k0 + 2 * BK);
        compute(1);
        if (notLast) {
            stage(0);
            __syncthreads();
        }
    }

    // ---- epilogue: bias (+swish); zero-fill pad cols; store f32/bf16 ----
#pragma unroll
    for (int u = 0; u < 4; ++u) {
        const int col = n0 + wn * 64 + u * 16 + lm;
        const bool colReal = col < N;
        const float bv = colReal ? bias[col] : 0.0f;
#pragma unroll
        for (int t = 0; t < 2; ++t) {
            const size_t rowBase = (size_t)(m0 + wm * 32 + t * 16 + lh * 8);
#pragma unroll
            for (int i = 0; i < 8; ++i) {
                float v = 0.0f;
                if (colReal) {
                    v = acc[t][u][i] + bv;
                    if (act) v = swishf(v);
                }
                const size_t idx = (rowBase + i) * Nstore + col;
                if (out_bf16) ((__bf16*)Cout)[idx] = (__bf16)v;
                else          ((float*)Cout)[idx]  = v;
            }
        }
    }
}

extern "C" void kernel_launch(void* const* d_in, const int* in_sizes, int n_in,
                              void* d_out, int out_size, void* d_ws,
                              size_t ws_size, hipStream_t stream) {
    const float* xa     = (const float*)d_in[0];
    const float* conv_w = (const float*)d_in[1];
    const float* conv_b = (const float*)d_in[2];
    const float* bn_g   = (const float*)d_in[3];
    const float* bn_b   = (const float*)d_in[4];
    const float* lin1_w = (const float*)d_in[5];
    const float* lin1_b = (const float*)d_in[6];
    const float* lin2_w = (const float*)d_in[7];
    const float* lin2_b = (const float*)d_in[8];
    const float* ln_g   = (const float*)d_in[9];
    const float* ln_b   = (const float*)d_in[10];
    const float* lin3_w = (const float*)d_in[11];
    const float* lin3_b = (const float*)d_in[12];
    const float* lin4_w = (const float*)d_in[13];
    const float* lin4_b = (const float*)d_in[14];

    char* ws = (char*)d_ws;
    size_t off = 0;
    auto carve = [&](size_t bytes) -> char* {
        char* p = ws + off;
        off += (bytes + 255) & ~(size_t)255;
        return p;
    };
    const int K1  = LOOPN * 16;  // 32768 (64-aligned)
    const int N1  = 1200, N1P = 1280;   // h1 padded width
    const int N2  = 741,  N2P = 768;    // h2/h2n padded width
    const int N3  = 240,  N3P = 256;    // h3 padded width
    const int N4  = 128;

    __bf16* z    = (__bf16*)carve((size_t)BATCHN * K1 * 2);    // 128 MB
    __bf16* w1b  = (__bf16*)carve((size_t)N1 * K1 * 2);        //  75 MB
    __bf16* w2b  = (__bf16*)carve((size_t)N2 * N1P * 2);
    __bf16* w3b  = (__bf16*)carve((size_t)N3 * N2P * 2);
    __bf16* w4b  = (__bf16*)carve((size_t)N4 * N3P * 2);
    float*  mean = (float*)carve((size_t)LOOPN * 4);
    float*  rstd = (float*)carve((size_t)LOOPN * 4);
    float*  bp   = (float*)carve((size_t)LOOPN * 4 * 4);
    __bf16* h1   = (__bf16*)carve((size_t)BATCHN * N1P * 2);
    float*  h2   = (float*)carve((size_t)BATCHN * N2P * 4);
    __bf16* h2n  = (__bf16*)carve((size_t)BATCHN * N2P * 2);
    __bf16* h3   = (__bf16*)carve((size_t)BATCHN * N3P * 2);

    // 1) per-channel conv+swish batch statistics
    conv_stats<<<LOOPN, 256, 0, stream>>>(xa, conv_w, conv_b, mean, rstd);
    // 2) 2x2 average pool broadcast factors
    pool_b<<<(LOOPN + 255) / 256, 256, 0, stream>>>(xa, bp);
    // 3) z = BN(swish(conv(x))) * bp, bf16 row-major [n][k]
    build_z<<<LOOPN, 256, 0, stream>>>(xa, conv_w, conv_b, bn_g, bn_b, mean,
                                       rstd, bp, z);
    // 4) weight conversions fp32 -> bf16, K zero-padded to GEMM-friendly lds
    cvt_pad<<<8192, 256, 0, stream>>>(lin1_w, w1b, N1, K1, K1);
    cvt_pad<<<1024, 256, 0, stream>>>(lin2_w, w2b, N2, N1, N1P);
    cvt_pad<<<512, 256, 0, stream>>>(lin3_w, w3b, N3, N2, N2P);
    cvt_pad<<<128, 256, 0, stream>>>(lin4_w, w4b, N4, N3, N3P);

    dim3 blk(256);
    const int mblks = BATCHN / BM;  // 16
    // 5) h1 = swish(z @ W1^T + b1)   (2048x32768) x (1200x32768)^T
    gemm_bf16<<<dim3(N1P / BN, mblks), blk, 0, stream>>>(
        z, w1b, lin1_b, h1, BATCHN, N1, N1P, K1, 1, 1);
    // 6) h2 = h1 @ W2^T + b2         (2048x1280) x (741x1280)^T
    gemm_bf16<<<dim3(N2P / BN, mblks), blk, 0, stream>>>(
        h1, w2b, lin2_b, h2, BATCHN, N2, N2P, N1P, 0, 0);
    // 7) layernorm(741) -> bf16 padded to 768
    ln_rows<<<BATCHN, 256, 0, stream>>>(h2, ln_g, ln_b, h2n, N2, N2P, N2P);
    // 8) h3 = swish(h2n @ W3^T + b3) (2048x768) x (240x768)^T
    gemm_bf16<<<dim3(N3P / BN, mblks), blk, 0, stream>>>(
        h2n, w3b, lin3_b, h3, BATCHN, N3, N3P, N2P, 1, 1);
    // 9) out = h3 @ W4^T + b4        (2048x256) x (128x256)^T -> f32 d_out
    gemm_bf16<<<dim3(N4 / BN, mblks), blk, 0, stream>>>(
        h3, w4b, lin4_b, d_out, BATCHN, N4, N4, N3P, 0, 0);
}
